// MultiHeadAttention_89163521065781
// MI455X (gfx1250) — matrix operand order repaired and checked by reading, not compiled, not run on hardware
//
#include <hip/hip_runtime.h>

// ---------------- constants ----------------
#define B_  8
#define S_  1024
#define E_  768
#define H_  12
#define DH_ 64
#define M_  (B_*S_)   // 8192 rows for the big GEMMs

// ---------------- types ----------------
typedef __bf16 bf16;
typedef __attribute__((ext_vector_type(16))) __bf16 v16bf;
typedef __attribute__((ext_vector_type(8)))  __bf16 v8bf;
typedef __attribute__((ext_vector_type(8)))  float  v8f;

union AFrag { v16bf v; v8bf h[2]; };

__device__ __forceinline__ bf16 f2bf(float f) {
  union { float f; unsigned u; } x; x.f = f;
  unsigned r = x.u + 0x7FFFu + ((x.u >> 16) & 1u);   // round-to-nearest-even
  unsigned short hs = (unsigned short)(r >> 16);
  union { unsigned short s; bf16 b; } y; y.s = hs;
  return y.b;
}

__device__ __forceinline__ v8f wmma_bf16(v16bf a, v16bf b, v8f c) {
  // D(16x16 f32) = A(16x32 bf16) * B(32x16 bf16) + C
  return __builtin_amdgcn_wmma_f32_16x16x32_bf16(
      false, a, false, b, (short)0, c, false, false);
}

// A fragment (16x32, row-major source, K contiguous).
// lanes 0-15 : M=lane,   K = 0..15
// lanes 16-31: M=lane-16, K = 16..31
__device__ __forceinline__ v16bf load_a_frag(const bf16* row, int half) {
  AFrag a;
  a.h[0] = *(const v8bf*)(row + half * 16);
  a.h[1] = *(const v8bf*)(row + half * 16 + 8);
  return a.v;
}

// B fragment (32x16): lane holds 16 contiguous K values for its column.
// lanes 0-15: K=0..15, lanes 16-31: K=16..31 (caller adds half*16 to K).
__device__ __forceinline__ v16bf load_b_frag(const bf16* p) {
  return *(const v16bf*)p;
}

// ---- CDNA5 async global->LDS copy (16B per lane), ASYNCcnt-tracked ----
// VDST = VGPR holding LDS byte address; VADDR = 64-bit global address.
__device__ __forceinline__ void async_ld16(unsigned lds_addr, const bf16* gptr) {
  asm volatile("global_load_async_to_lds_b128 %0, %1, off"
               :: "v"(lds_addr),
                  "v"((unsigned long long)(uintptr_t)gptr)
               : "memory");
}

template <int N>
__device__ __forceinline__ void wait_async() {
#if defined(__has_builtin)
#if __has_builtin(__builtin_amdgcn_s_wait_asynccnt)
  __builtin_amdgcn_s_wait_asynccnt(N);
  return;
#endif
#endif
  asm volatile("s_wait_asynccnt %0" :: "i"(N) : "memory");
}

// Low 32 bits of a flat LDS pointer == LDS byte offset (flat->LDS truncates).
__device__ __forceinline__ unsigned lds_addr_of(const void* p) {
  return (unsigned)(uintptr_t)p;
}

// ---------------- f32 -> bf16 conversion ----------------
__global__ void cvt_f32_bf16(const float* __restrict__ in,
                             bf16* __restrict__ out, int n) {
  int i = blockIdx.x * blockDim.x + threadIdx.x;
  if (i < n) out[i] = f2bf(in[i]);
}

// ---------------- QKV projection GEMM ----------------
// C[M,N] = A[M,K] * W[N,K]^T + bias[N], A=[8192,768] bf16, W=[768,768] bf16.
// Weight tile (64 cols x 32 K = 4KB) staged block-wide into a DOUBLE-BUFFERED
// LDS tile with global_load_async_to_lds_b128: iteration i prefetches tile i+1
// while WMMAs consume tile i (wait asynccnt<=1 = tile i complete, in-order).
// mode 0/1: store [B,H,S,DH] (Q, K).  mode 2: store [B,H,DH,S] (V transposed).
__global__ __launch_bounds__(256)
void proj_kernel(const bf16* __restrict__ X, const bf16* __restrict__ W,
                 const float* __restrict__ bias, bf16* __restrict__ out,
                 int mode) {
  const int tid  = threadIdx.x;
  const int lane = tid & 31;
  const int wave = tid >> 5;
  const int half = lane >> 4;
  const int lcol = lane & 15;
  const int m0 = blockIdx.x * 128 + wave * 16;
  const int n0 = blockIdx.y * 64;

  __shared__ bf16 Wlds[2][64 * 32];       // double-buffered weight tile (2x4KB)
  const int wrow   = tid >> 2;            // 0..63  (n offset)
  const int wchunk = tid & 3;             // 0..3   (8 bf16 = 16B each)
  unsigned lds_dst[2];
  lds_dst[0] = lds_addr_of(&Wlds[0][wrow * 32 + wchunk * 8]);
  lds_dst[1] = lds_addr_of(&Wlds[1][wrow * 32 + wchunk * 8]);
  const bf16* wsrc_base = W + (size_t)(n0 + wrow) * E_ + wchunk * 8;

  v8f acc[4] = {};
  const bf16* arow_base = X + (size_t)(m0 + lcol) * E_;

  async_ld16(lds_dst[0], wsrc_base);      // prologue: stage first tile
  int cur = 0;
  for (int k0 = 0; k0 < E_; k0 += 32) {
    if (k0 + 32 < E_) {                   // prefetch next tile into other buffer
      async_ld16(lds_dst[cur ^ 1], wsrc_base + k0 + 32);
      wait_async<1>();                    // current tile done (in-order returns)
    } else {
      wait_async<0>();
    }
    __syncthreads();                      // tile complete across all waves

    v16bf a = load_a_frag(arow_base + k0, half);
    const bf16* wl = &Wlds[cur][0];
#pragma unroll
    for (int t = 0; t < 4; ++t) {
      const bf16* bp = wl + (t * 16 + lcol) * 32 + half * 16;
      acc[t] = wmma_bf16(a, load_b_frag(bp), acc[t]);
    }
    __syncthreads();                      // reads done before buffer is re-filled
    cur ^= 1;
  }

#pragma unroll
  for (int t = 0; t < 4; ++t) {
    const int n = n0 + t * 16 + lcol;
    const float bv = bias[n];
    const int hh = n >> 6;      // head
    const int d  = n & 63;      // dim in head
#pragma unroll
    for (int r = 0; r < 8; ++r) {
      const int m = m0 + r + half * 8;
      const int b = m >> 10;
      const int s = m & 1023;
      size_t addr;
      if (mode < 2) addr = ((size_t)(b * H_ + hh) * S_ + s) * DH_ + d;
      else          addr = ((size_t)(b * H_ + hh) * DH_ + d) * S_ + s;
      out[addr] = f2bf(acc[t][r] + bv);
    }
  }
}

// ---------------- attention (flash-style, one wave / 16 q-rows) ----------------
__global__ __launch_bounds__(32)
void attn_kernel(const bf16* __restrict__ Q, const bf16* __restrict__ K,
                 const bf16* __restrict__ VT, bf16* __restrict__ cat) {
  const int q0 = blockIdx.x * 16;
  const int h  = blockIdx.y;
  const int b  = blockIdx.z;
  const int lane = threadIdx.x & 31;
  const int half = lane >> 4;
  const int lcol = lane & 15;

  const size_t bh = (size_t)(b * H_ + h);
  const bf16* Qb = Q  + bh * S_ * DH_;
  const bf16* Kb = K  + bh * S_ * DH_;
  const bf16* Vb = VT + bh * DH_ * S_;

  // Q fragments for the two 32-wide K-dim steps over DH=64, held in registers
  v16bf aq[2];
  {
    const bf16* qrow = Qb + (size_t)(q0 + lcol) * DH_;
    aq[0] = load_a_frag(qrow +  0, half);
    aq[1] = load_a_frag(qrow + 32, half);
  }

  float mrow[8], lrow[8];
  v8f acc[4] = {};
#pragma unroll
  for (int r = 0; r < 8; ++r) { mrow[r] = -3.0e38f; lrow[r] = 0.f; }

  __shared__ bf16 P[16 * 32];
  const float scale = 0.125f;  // 1/sqrt(64)

  for (int t0 = 0; t0 < S_; t0 += 32) {
    // scores: two 16x16 tiles (keys t0..+15 and t0+16..+31)
    v8f sc0 = {}, sc1 = {};
#pragma unroll
    for (int ks = 0; ks < 2; ++ks) {
      const bf16* kp0 = Kb + (size_t)(t0 + lcol) * DH_ + ks * 32 + half * 16;
      sc0 = wmma_bf16(aq[ks], load_b_frag(kp0), sc0);
      const bf16* kp1 = Kb + (size_t)(t0 + 16 + lcol) * DH_ + ks * 32 + half * 16;
      sc1 = wmma_bf16(aq[ks], load_b_frag(kp1), sc1);
    }

    // online softmax (row = vgpr slot, columns across the 16-lane group)
    float pr0[8], pr1[8];
#pragma unroll
    for (int r = 0; r < 8; ++r) {
      float s0 = sc0[r] * scale;
      float s1 = sc1[r] * scale;
      float mx = fmaxf(s0, s1);
#pragma unroll
      for (int off = 1; off < 16; off <<= 1)
        mx = fmaxf(mx, __shfl_xor(mx, off, 32));
      const float mnew  = fmaxf(mrow[r], mx);
      const float alpha = __expf(mrow[r] - mnew);
      const float p0 = __expf(s0 - mnew);
      const float p1 = __expf(s1 - mnew);
      float rs = p0 + p1;
#pragma unroll
      for (int off = 1; off < 16; off <<= 1)
        rs += __shfl_xor(rs, off, 32);
      lrow[r] = lrow[r] * alpha + rs;
      mrow[r] = mnew;
      pr0[r] = p0; pr1[r] = p1;
#pragma unroll
      for (int t = 0; t < 4; ++t) acc[t][r] *= alpha;
    }

    // C-layout -> A-layout via LDS (single wave: DS ops are in-order)
#pragma unroll
    for (int r = 0; r < 8; ++r) {
      const int row = r + half * 8;
      P[row * 32 + lcol]      = f2bf(pr0[r]);
      P[row * 32 + 16 + lcol] = f2bf(pr1[r]);
    }
    __builtin_amdgcn_wave_barrier();
    v16bf ap = load_a_frag(&P[lcol * 32], half);
    __builtin_amdgcn_wave_barrier();

    // attn . V : 4 WMMAs (N = dh tiles), Vt rows are contiguous in keys
#pragma unroll
    for (int t = 0; t < 4; ++t) {
      const bf16* vp = Vb + (size_t)(t * 16 + lcol) * S_ + t0 + half * 16;
      acc[t] = wmma_bf16(ap, load_b_frag(vp), acc[t]);
    }
  }

  // epilogue: normalize and store to cat[B,S,E] bf16
#pragma unroll
  for (int t = 0; t < 4; ++t) {
    const int d = t * 16 + lcol;
#pragma unroll
    for (int r = 0; r < 8; ++r) {
      const int row = r + half * 8;
      const int s = q0 + row;
      const float v = acc[t][r] / lrow[r];
      cat[((size_t)(b * S_ + s)) * E_ + h * DH_ + d] = f2bf(v);
    }
  }
}

// ---------------- output projection GEMM (f32 store) ----------------
__global__ __launch_bounds__(256)
void outproj_kernel(const bf16* __restrict__ X, const bf16* __restrict__ W,
                    const float* __restrict__ bias, float* __restrict__ out) {
  const int tid  = threadIdx.x;
  const int lane = tid & 31;
  const int wave = tid >> 5;
  const int half = lane >> 4;
  const int lcol = lane & 15;
  const int m0 = blockIdx.x * 128 + wave * 16;
  const int n0 = blockIdx.y * 64;

  __shared__ bf16 Wlds[2][64 * 32];
  const int wrow   = tid >> 2;
  const int wchunk = tid & 3;
  unsigned lds_dst[2];
  lds_dst[0] = lds_addr_of(&Wlds[0][wrow * 32 + wchunk * 8]);
  lds_dst[1] = lds_addr_of(&Wlds[1][wrow * 32 + wchunk * 8]);
  const bf16* wsrc_base = W + (size_t)(n0 + wrow) * E_ + wchunk * 8;

  v8f acc[4] = {};
  const bf16* arow_base = X + (size_t)(m0 + lcol) * E_;

  async_ld16(lds_dst[0], wsrc_base);
  int cur = 0;
  for (int k0 = 0; k0 < E_; k0 += 32) {
    if (k0 + 32 < E_) {
      async_ld16(lds_dst[cur ^ 1], wsrc_base + k0 + 32);
      wait_async<1>();
    } else {
      wait_async<0>();
    }
    __syncthreads();

    v16bf a = load_a_frag(arow_base + k0, half);
    const bf16* wl = &Wlds[cur][0];
#pragma unroll
    for (int t = 0; t < 4; ++t) {
      const bf16* bp = wl + (t * 16 + lcol) * 32 + half * 16;
      acc[t] = wmma_bf16(a, load_b_frag(bp), acc[t]);
    }
    __syncthreads();
    cur ^= 1;
  }

#pragma unroll
  for (int t = 0; t < 4; ++t) {
    const int n = n0 + t * 16 + lcol;
    const float bv = bias[n];
#pragma unroll
    for (int r = 0; r < 8; ++r) {
      const int m = m0 + r + half * 8;
      out[(size_t)m * E_ + n] = acc[t][r] + bv;
    }
  }
}

// ---------------- host launch ----------------
extern "C" void kernel_launch(void* const* d_in, const int* in_sizes, int n_in,
                              void* d_out, int out_size, void* d_ws, size_t ws_size,
                              hipStream_t stream) {
  const float* x  = (const float*)d_in[0];
  const float* Wq = (const float*)d_in[1];
  const float* bq = (const float*)d_in[2];
  const float* Wk = (const float*)d_in[3];
  const float* bk = (const float*)d_in[4];
  const float* Wv = (const float*)d_in[5];
  const float* bv = (const float*)d_in[6];
  const float* Wo = (const float*)d_in[7];
  const float* bo = (const float*)d_in[8];
  float* out = (float*)d_out;

  // workspace carve-out (bf16 buffers), 256B aligned
  char* base = (char*)d_ws;
  size_t off = 0;
  auto carve = [&](size_t elems) {
    char* p = base + off;
    off = (off + elems * sizeof(bf16) + 255) & ~(size_t)255;
    return (bf16*)p;
  };
  const size_t nx = (size_t)B_ * S_ * E_;   // 6,291,456
  const size_t nw = (size_t)E_ * E_;        //   589,824
  bf16* xb  = carve(nx);
  bf16* Wqb = carve(nw);
  bf16* Wkb = carve(nw);
  bf16* Wvb = carve(nw);
  bf16* Wob = carve(nw);
  bf16* Qb  = carve(nx);   // [B,H,S,DH]
  bf16* Kb  = carve(nx);   // [B,H,S,DH]
  bf16* Vt  = carve(nx);   // [B,H,DH,S]
  bf16* cat = carve(nx);   // [B,S,E]
  (void)ws_size; (void)in_sizes; (void)n_in; (void)out_size;

  // 1) f32 -> bf16
  cvt_f32_bf16<<<(int)((nx + 255) / 256), 256, 0, stream>>>(x,  xb,  (int)nx);
  cvt_f32_bf16<<<(int)((nw + 255) / 256), 256, 0, stream>>>(Wq, Wqb, (int)nw);
  cvt_f32_bf16<<<(int)((nw + 255) / 256), 256, 0, stream>>>(Wk, Wkb, (int)nw);
  cvt_f32_bf16<<<(int)((nw + 255) / 256), 256, 0, stream>>>(Wv, Wvb, (int)nw);
  cvt_f32_bf16<<<(int)((nw + 255) / 256), 256, 0, stream>>>(Wo, Wob, (int)nw);

  // 2) Q/K/V projections (WMMA GEMM, M=8192 N=768 K=768, async double-buffer)
  dim3 gGemm(M_ / 128, E_ / 64);
  proj_kernel<<<gGemm, 256, 0, stream>>>(xb, Wqb, bq, Qb, 0);
  proj_kernel<<<gGemm, 256, 0, stream>>>(xb, Wkb, bk, Kb, 1);
  proj_kernel<<<gGemm, 256, 0, stream>>>(xb, Wvb, bv, Vt, 2);

  // 3) attention: one wave per (b, h, 16-query rows)
  dim3 gAttn(S_ / 16, H_, B_);
  attn_kernel<<<gAttn, 32, 0, stream>>>(Qb, Kb, Vt, cat);

  // 4) output projection (f32 result + bias)
  outproj_kernel<<<gGemm, 256, 0, stream>>>(cat, Wob, bo, out);
}
